// DPLLM_Linear_Finetune_65850438582287
// MI455X (gfx1250) — matrix operand order repaired and checked by reading, not compiled, not run on hardware
//
#include <hip/hip_runtime.h>

typedef __attribute__((ext_vector_type(16))) __bf16 v16bf;
typedef __attribute__((ext_vector_type(8)))  __bf16 v8bf;
typedef __attribute__((ext_vector_type(2)))  __bf16 v2bf;
typedef __attribute__((ext_vector_type(8)))  float  v8f;
typedef __attribute__((ext_vector_type(4)))  float  v4f;
typedef __attribute__((ext_vector_type(4)))  unsigned u32x4;
typedef __attribute__((ext_vector_type(8)))  unsigned u32x8;

#define IN_F   4096
#define OUT_F  4096
#define M_TOT  8192      // 4 * 2048
#define BM     128
#define BN     128
#define BK     32
#define LDK    (BK + 8)                 // 80B LDS row stride -> conflict-free b128 frag loads
#define WS_TILE_BYTES (BN * LDK * 2)    // 10240 B per W LDS buffer

// f32 -> packed bf16 pair; native v_cvt_pk_bf16_f32 when available.
static __device__ __forceinline__ v2bf pk2bf(float a, float b) {
#if __has_builtin(__builtin_amdgcn_cvt_pk_bf16_f32)
    return __builtin_amdgcn_cvt_pk_bf16_f32(a, b);
#else
    v2bf r;
    r[0] = (__bf16)a;
    r[1] = (__bf16)b;
    return r;
#endif
}

// ---------------------------------------------------------------------------
// Kernel 1: fused bitplane-LUT dequant + precision blend.
//   W_eff[o][i] = r * lut4[o][code4] + (1-r) * lut5[o][code5]   (stored bf16)
// ---------------------------------------------------------------------------
__global__ __launch_bounds__(256)
void dequant_blend_kernel(const int* __restrict__ qw,
                          const float* __restrict__ lut4,
                          const float* __restrict__ lut5,
                          const float* __restrict__ zp,
                          unsigned short* __restrict__ Wout) {
    const int g  = blockIdx.x * blockDim.x + threadIdx.x;  // 0 .. OUT_F*(IN_F/32)-1
    const int o  = g >> 7;
    const int gc = g & 127;
    // p = sigmoid(z)*3 + 3 ; r = 1 - (p - 4) = 2 - 3*sigmoid(z)
    const float sig = 1.0f / (1.0f + __expf(-zp[0]));
    const float r   = 2.0f - 3.0f * sig;
    const float rb  = 1.0f - r;

    const int plane_stride = OUT_F * (IN_F / 32);
    const int base = o * (IN_F / 32) + gc;
    const int q0 = qw[0 * plane_stride + base];
    const int q1 = qw[1 * plane_stride + base];
    const int q2 = qw[2 * plane_stride + base];
    const int q3 = qw[3 * plane_stride + base];
    const int q4 = qw[4 * plane_stride + base];
    const float* l4 = lut4 + o * 16;
    const float* l5 = lut5 + o * 32;

    union { v2bf p[16]; uint4 v[4]; } wv;
    #pragma unroll
    for (int t = 0; t < 32; t += 2) {
        const int c4a = (((q0 >> t) & 1) << 3) | (((q1 >> t) & 1) << 2) |
                        (((q2 >> t) & 1) << 1) |  ((q3 >> t) & 1);
        const int c5a = (c4a << 1) | ((q4 >> t) & 1);
        const int t1  = t + 1;
        const int c4b = (((q0 >> t1) & 1) << 3) | (((q1 >> t1) & 1) << 2) |
                        (((q2 >> t1) & 1) << 1) |  ((q3 >> t1) & 1);
        const int c5b = (c4b << 1) | ((q4 >> t1) & 1);
        const float wa = r * l4[c4a] + rb * l5[c5a];
        const float wb = r * l4[c4b] + rb * l5[c5b];
        wv.p[t >> 1] = pk2bf(wa, wb);
    }
    uint4* dst = (uint4*)(Wout + (size_t)o * IN_F + (size_t)gc * 32);
    dst[0] = wv.v[0]; dst[1] = wv.v[1]; dst[2] = wv.v[2]; dst[3] = wv.v[3];
}

// ---------------------------------------------------------------------------
// TDM: DMA one 128x32 bf16 W tile (row stride IN_F elements) into LDS, with
// hardware padding of 4 DWORDs after every 16 DWORDs -> 80B LDS row stride,
// matching the LDK=40 layout the WMMA fragment loads expect.
// D# per CDNA5 ISA. Issued by one wave; TENSORcnt-tracked.
// ---------------------------------------------------------------------------
static __device__ __forceinline__ void tdm_load_w_tile(const unsigned short* Wg,
                                                       unsigned lds_byte_off,
                                                       int N0, int k0) {
    const unsigned long long ga =
        (unsigned long long)(const void*)(Wg + (size_t)N0 * IN_F + (size_t)k0);
    u32x4 g0;
    g0[0] = 0x1u;                                  // count=1, user descriptor
    g0[1] = lds_byte_off;                          // lds_addr
    g0[2] = (unsigned)ga;                          // global_addr[31:0]
    g0[3] = (unsigned)(ga >> 32) | (2u << 30);     // global_addr[56:32] | type=2
    u32x8 g1;
    g1[0] = (1u << 16)                             // data_size = 2 bytes
          | (1u << 20)                             // pad_enable
          | (3u << 22)                             // pad_interval: 16 DWORDs (64B = one row)
          | (3u << 25);                            // pad_amount: 4 DWORDs (16B)
    g1[1] = ((unsigned)IN_F  & 0xffffu) << 16;     // tensor_dim0[15:0]
    g1[2] = ((unsigned)OUT_F & 0xffffu) << 16;     // tensor_dim0 hi=0 | tensor_dim1[15:0]
    g1[3] = 32u << 16;                             // tensor_dim1 hi=0 | tile_dim0 = 32
    g1[4] = 128u;                                  // tile_dim1 = 128, tile_dim2 = 0
    g1[5] = (unsigned)IN_F;                        // tensor_dim0_stride[31:0]
    g1[6] = 0u;
    g1[7] = 0u;
    asm volatile("tensor_load_to_lds %0, %1" :: "s"(g0), "s"(g1) : "memory");
}

// ---------------------------------------------------------------------------
// Kernel 2: Y = X * W_eff^T + bias via v_wmma_f32_16x16x32_bf16.
// 128x128 block tile, 8 waves (2Mx4N), wave tile 64x32 = 4x2 WMMA tiles.
// Double-buffered LDS; W tiles via TDM, X tiles converted f32->bf16 with a
// register pipeline; single barrier per K-step.
// ---------------------------------------------------------------------------
__global__ __launch_bounds__(256)
void wmma_gemm_bias_kernel(const float* __restrict__ X,
                           const unsigned short* __restrict__ Wu,
                           const float* __restrict__ bias,
                           float* __restrict__ Y) {
    __shared__ __align__(128) __bf16 Xs[2][BM][LDK];
    __shared__ __align__(128) __bf16 Ws[2][BN][LDK];

    const int tid  = threadIdx.x;
    const int lane = tid & 31;
    const int wave = tid >> 5;            // 0..7
    const int M0 = blockIdx.y * BM;
    const int N0 = blockIdx.x * BN;
    const int wm = (wave >> 2) * 64;      // 0 or 64
    const int wn = (wave & 3) * 32;       // 0,32,64,96

    const int lrow  = lane & 15;
    const int lhalf = lane >> 4;          // 0 or 1

    const int srow   = tid >> 1;          // 0..127
    const int schunk = (tid & 1) * 16;    // 0 or 16

    const unsigned ws_base = (unsigned)(unsigned long long)
        (__attribute__((address_space(3))) const void*)&Ws[0][0][0];

    v8f acc[4][2] = {};

    // ---- prologue: TDM W tile 0, preload X tile 0 into registers ----
    if (wave == 0) tdm_load_w_tile(Wu, ws_base, N0, 0);

    const float* xsrc = X + (size_t)(M0 + srow) * IN_F + schunk;
    v4f x0 = *(const v4f*)(xsrc + 0);
    v4f x1 = *(const v4f*)(xsrc + 4);
    v4f x2 = *(const v4f*)(xsrc + 8);
    v4f x3 = *(const v4f*)(xsrc + 12);

    for (int k0 = 0, it = 0; k0 < IN_F; k0 += BK, ++it) {
        const int cur = it & 1;
        const int nxt = cur ^ 1;
        const bool more = (k0 + BK) < IN_F;

        // ---- stage X tile (regs -> bf16 via v_cvt_pk_bf16_f32 -> LDS) ----
        {
            union { v2bf p[8]; v8bf h[2]; } t;
            t.p[0] = pk2bf(x0[0], x0[1]);
            t.p[1] = pk2bf(x0[2], x0[3]);
            t.p[2] = pk2bf(x1[0], x1[1]);
            t.p[3] = pk2bf(x1[2], x1[3]);
            t.p[4] = pk2bf(x2[0], x2[1]);
            t.p[5] = pk2bf(x2[2], x2[3]);
            t.p[6] = pk2bf(x3[0], x3[1]);
            t.p[7] = pk2bf(x3[2], x3[3]);
            v8bf* d = (v8bf*)&Xs[cur][srow][schunk];
            d[0] = t.h[0];
            d[1] = t.h[1];
        }

        // Ws[cur] must be complete before anyone passes the barrier.
        if (wave == 0) __builtin_amdgcn_s_wait_tensorcnt(0);
        __syncthreads();

        // ---- overlap next tile's data movement with this tile's math ----
        if (more) {
            if (wave == 0)
                tdm_load_w_tile(Wu, ws_base + (unsigned)nxt * WS_TILE_BYTES, N0, k0 + BK);
            const float* xn = xsrc + k0 + BK;
            x0 = *(const v4f*)(xn + 0);
            x1 = *(const v4f*)(xn + 4);
            x2 = *(const v4f*)(xn + 8);
            x3 = *(const v4f*)(xn + 12);
        }

        // ---- fragments (ISA 16-bit A 16x32 / B 32x16 layouts) ----
        v16bf a[4];
        #pragma unroll
        for (int mt = 0; mt < 4; ++mt) {
            const __bf16* p = &Xs[cur][wm + mt * 16 + lrow][lhalf * 8];
            union { v16bf v; v8bf h[2]; } u;
            u.h[0] = *(const v8bf*)p;         // K = h*8 .. h*8+7
            u.h[1] = *(const v8bf*)(p + 16);  // K = h*8+16 .. h*8+23
            a[mt] = u.v;
        }
        v16bf b[2];
        #pragma unroll
        for (int nt = 0; nt < 2; ++nt) {
            const __bf16* p = &Ws[cur][wn + nt * 16 + lrow][lhalf * 8];
            union { v16bf v; v8bf h[2]; } u;
            u.h[0] = *(const v8bf*)p;         // K = h*8 .. h*8+7
            u.h[1] = *(const v8bf*)(p + 16);  // K = h*8+16 .. h*8+23
            b[nt] = u.v;
        }

        #pragma unroll
        for (int mt = 0; mt < 4; ++mt)
            #pragma unroll
            for (int nt = 0; nt < 2; ++nt)
                acc[mt][nt] = __builtin_amdgcn_wmma_f32_16x16x32_bf16(
                    false, a[mt], false, b[nt], (short)0, acc[mt][nt], false, false);
        // no trailing barrier: next iteration writes the other LDS buffer, and
        // the next __syncthreads (which waits DScnt) orders buffer reuse.
    }

    // ---- epilogue: D VGPR r -> row (r + lhalf*8), col = lane%16; add bias ----
    #pragma unroll
    for (int nt = 0; nt < 2; ++nt) {
        const int col = N0 + wn + nt * 16 + lrow;
        const float bv = bias[col];
        #pragma unroll
        for (int mt = 0; mt < 4; ++mt) {
            #pragma unroll
            for (int rr = 0; rr < 8; ++rr) {
                const int row = M0 + wm + mt * 16 + lhalf * 8 + rr;
                Y[(size_t)row * OUT_F + col] = acc[mt][nt][rr] + bv;
            }
        }
    }
}

// ---------------------------------------------------------------------------
extern "C" void kernel_launch(void* const* d_in, const int* in_sizes, int n_in,
                              void* d_out, int out_size, void* d_ws, size_t ws_size,
                              hipStream_t stream) {
    (void)in_sizes; (void)n_in; (void)out_size; (void)ws_size;
    const float* x    = (const float*)d_in[0];   // (4,2048,4096) f32
    const float* z    = (const float*)d_in[1];   // scalar
    const float* lut4 = (const float*)d_in[2];   // (4096,16)
    const float* lut5 = (const float*)d_in[3];   // (4096,32)
    const float* bias = (const float*)d_in[4];   // (4096,)
    const int*   qw   = (const int*)d_in[5];     // (6,4096,128) int32
    float* out = (float*)d_out;                  // (4,2048,4096) f32

    unsigned short* Weff = (unsigned short*)d_ws;  // 4096*4096 bf16 = 32 MB

    dequant_blend_kernel<<<(OUT_F * (IN_F / 32)) / 256, 256, 0, stream>>>(
        qw, lut4, lut5, z, Weff);

    dim3 grid(OUT_F / BN, M_TOT / BM);   // (32, 64), N-fastest for L2 reuse of X
    wmma_gemm_bias_kernel<<<grid, 256, 0, stream>>>(x, Weff, bias, out);
}
